// MultiHeadAttention_34961033790236
// MI455X (gfx1250) — matrix operand order repaired and checked by reading, not compiled, not run on hardware
//
#include <hip/hip_runtime.h>

typedef __attribute__((ext_vector_type(16))) __bf16 v16bf;
typedef __attribute__((ext_vector_type(8)))  __bf16 v8bf;
typedef __attribute__((ext_vector_type(8)))  float  v8f;
typedef __attribute__((ext_vector_type(4)))  float  f4;
typedef __attribute__((ext_vector_type(4)))  unsigned short u16x4;

#define D_MODEL 1024
#define S_LEN   2048
#define NB      4
#define NH      16
#define DH      64

// float -> bf16 bits, round-to-nearest-even
__device__ __forceinline__ unsigned short f2bfu(float f) {
    union { float f; unsigned u; } c; c.f = f;
    unsigned u = c.u + 0x7FFFu + ((c.u >> 16) & 1u);
    return (unsigned short)(u >> 16);
}

__device__ __forceinline__ v16bf cat8(v8bf lo, v8bf hi) {
    return __builtin_shufflevector(lo, hi, 0,1,2,3,4,5,6,7,8,9,10,11,12,13,14,15);
}

__device__ __forceinline__ v8f wmma_bf16(v16bf a, v16bf b, v8f c) {
    return __builtin_amdgcn_wmma_f32_16x16x32_bf16(
        false, a, false, b, (short)0, c, false, false);
}

// ---------------------------------------------------------------------------
// Elementwise f32 -> bf16 conversion (x and the four weight matrices).
// ---------------------------------------------------------------------------
__global__ void __launch_bounds__(256)
cvt_bf16_kernel(const float* __restrict__ src,
                unsigned short* __restrict__ dst, int n4) {
    int i = blockIdx.x * blockDim.x + threadIdx.x;
    if (i >= n4) return;
    f4 v = *(const f4*)(src + (size_t)i * 4);
    u16x4 o;
    o.x = f2bfu(v.x); o.y = f2bfu(v.y); o.z = f2bfu(v.z); o.w = f2bfu(v.w);
    *(u16x4*)(dst + (size_t)i * 4) = o;
}

// A tiles: 2 row-subtiles of 16 rows; per-lane two 16B chunks (K = g*16+e)
__device__ __forceinline__ void load_a2(v16bf A[2], const unsigned short* base,
                                        int d0, int g) {
#pragma unroll
    for (int m = 0; m < 2; ++m) {
        const unsigned short* p = base + (size_t)m * 16 * D_MODEL + d0;
        A[m] = cat8(*(const v8bf*)(p + g * 16), *(const v8bf*)(p + g * 16 + 8));
    }
}

// B tiles: 4 col-subtiles of 16 cols; per-lane one 32B chunk (K = g*16+e)
__device__ __forceinline__ void load_b4(v16bf B[4], const unsigned short* base,
                                        int d0) {
#pragma unroll
    for (int j = 0; j < 4; ++j)
        B[j] = *(const v16bf*)(base + (size_t)j * 16 * D_MODEL + d0);
}

__device__ __forceinline__ void mma8(v8f acc[2][4], const v16bf A[2], const v16bf B[4]) {
#pragma unroll
    for (int m = 0; m < 2; ++m)
#pragma unroll
        for (int j = 0; j < 4; ++j)
            acc[m][j] = wmma_bf16(A[m], B[j], acc[m][j]);
}

// ---------------------------------------------------------------------------
// Projection: out = x @ W^T + b. One wave computes a 32-row x 64-col (one
// head) tile, double-buffered over K. ~190 VGPRs -> no spills under
// __launch_bounds__(256).
//   mode 0: Q -> scale by 0.125, store [b,h,s,d]
//   mode 1: K -> store [b,h,s,d]
//   mode 2: V -> store transposed [b,h,d,s]
// ---------------------------------------------------------------------------
__global__ void __launch_bounds__(256)
proj_kernel(const unsigned short* __restrict__ xbf,
            const unsigned short* __restrict__ Wbf,
            const float* __restrict__ bias,
            unsigned short* __restrict__ outw,
            int mode) {
    int tid = blockIdx.x * blockDim.x + threadIdx.x;
    int w = tid >> 5;
    int lane = tid & 31, lane16 = lane & 15, g = lane >> 4;
    int mtile = w >> 4;          // 0..255 (32-row tiles of flattened [B*S])
    int head  = w & 15;

    const unsigned short* abase = xbf + (size_t)(mtile * 32 + lane16) * D_MODEL;
    const unsigned short* bbase = Wbf + (size_t)(head * 64 + lane16) * D_MODEL + g * 16;

    v8f z = {0.f,0.f,0.f,0.f,0.f,0.f,0.f,0.f};
    v8f acc[2][4];
#pragma unroll
    for (int m = 0; m < 2; ++m)
#pragma unroll
        for (int j = 0; j < 4; ++j) acc[m][j] = z;

    v16bf A0[2], B0[4], A1[2], B1[4];
    load_a2(A0, abase, 0, g);
    load_b4(B0, bbase, 0);
    for (int d0 = 0; d0 < D_MODEL; d0 += 64) {
        load_a2(A1, abase, d0 + 32, g);
        load_b4(B1, bbase, d0 + 32);
        mma8(acc, A0, B0);
        if (d0 + 64 < D_MODEL) {
            load_a2(A0, abase, d0 + 64, g);
            load_b4(B0, bbase, d0 + 64);
        }
        mma8(acc, A1, B1);
    }

#pragma unroll
    for (int j = 0; j < 4; ++j) {
        int col = j * 16 + lane16;            // within head (0..63)
        float bv = bias[head * 64 + col];
#pragma unroll
        for (int m = 0; m < 2; ++m)
#pragma unroll
            for (int r = 0; r < 8; ++r) {
                int row = mtile * 32 + m * 16 + r + 8 * g;   // flattened B*S
                int b = row >> 11, s = row & (S_LEN - 1);
                float v = acc[m][j][r] + bv;
                if (mode == 0) v *= 0.125f;   // 1/sqrt(Dh)
                size_t idx;
                if (mode == 2)
                    idx = ((size_t)((b * NH + head) * DH + col)) * S_LEN + s;
                else
                    idx = ((size_t)((b * NH + head) * S_LEN + s)) * DH + col;
                outw[idx] = f2bfu(v);
            }
    }
}

// ---------------------------------------------------------------------------
// Flash attention: one wave = (b, h, 32-query tile). Online softmax, causal
// mask analytic. Liveness phased to stay under 256 VGPRs: KB dies after the
// score WMMAs, then VB is issued (in flight across the softmax VALU chain).
// P transposed C->A layout via per-wave LDS (DS ops in-order per wave).
// ---------------------------------------------------------------------------
__global__ void __launch_bounds__(256)
attn_kernel(const unsigned short* __restrict__ qws,
            const unsigned short* __restrict__ kws,
            const unsigned short* __restrict__ vtws,
            unsigned short* __restrict__ ctxws) {
    __shared__ unsigned short ptile[8][32 * 40];   // per-wave 32x32 bf16, stride 40

    int tid = blockIdx.x * blockDim.x + threadIdx.x;
    int w = tid >> 5;
    int wid = threadIdx.x >> 5;
    int lane = tid & 31, lane16 = lane & 15, g = lane >> 4;
    int qt = w & 63;             // 32-query tile within head
    int bh = w >> 6;             // b*16 + h
    int qbase = qt * 32;

    const unsigned short* qp = qws  + (size_t)bh * S_LEN * DH;
    const unsigned short* kp = kws  + (size_t)bh * S_LEN * DH;
    const unsigned short* vp = vtws + (size_t)bh * DH * S_LEN;

    // Q A-tiles held for the whole loop: [m-subtile][d-chunk]
    v16bf AQ[2][2];
#pragma unroll
    for (int m = 0; m < 2; ++m) {
        const unsigned short* qrow = qp + (size_t)(qbase + m * 16 + lane16) * DH;
        AQ[m][0] = cat8(*(const v8bf*)(qrow + g * 16),      *(const v8bf*)(qrow + g * 16 + 8));
        AQ[m][1] = cat8(*(const v8bf*)(qrow + 32 + g * 16), *(const v8bf*)(qrow + 40 + g * 16));
    }

    v8f z = {0.f,0.f,0.f,0.f,0.f,0.f,0.f,0.f};
    v8f acc[2][4];
    float m_run[2][8], l_run[2][8];
#pragma unroll
    for (int m = 0; m < 2; ++m) {
#pragma unroll
        for (int j = 0; j < 4; ++j) acc[m][j] = z;
#pragma unroll
        for (int r = 0; r < 8; ++r) { m_run[m][r] = -3.0e38f; l_run[m][r] = 0.f; }
    }

    unsigned short* pw = &ptile[wid][0];

    for (int k0 = 0; k0 < qbase + 32; k0 += 32) {
        // ---- phase 1: K tiles + score WMMAs (KB live) ----
        v8f s[2][2];
        {
            v16bf KB[2][2];
#pragma unroll
            for (int nt = 0; nt < 2; ++nt) {
                const unsigned short* kr = kp + (size_t)(k0 + nt * 16 + lane16) * DH;
                KB[nt][0] = *(const v16bf*)(kr + g * 16);
                KB[nt][1] = *(const v16bf*)(kr + 32 + g * 16);
            }
#pragma unroll
            for (int m = 0; m < 2; ++m)
#pragma unroll
                for (int nt = 0; nt < 2; ++nt) {
                    v8f t = wmma_bf16(AQ[m][0], KB[nt][0], z);
                    s[m][nt] = wmma_bf16(AQ[m][1], KB[nt][1], t);
                }
        }

        // ---- phase 2: V loads in flight while softmax runs (VB replaces KB) ----
        v16bf VB[4];
#pragma unroll
        for (int j = 0; j < 4; ++j)
            VB[j] = *(const v16bf*)(vp + (size_t)(j * 16 + lane16) * S_LEN + k0 + g * 16);

#pragma unroll
        for (int m = 0; m < 2; ++m) {
            int rowbase = qbase + m * 16;
            if (k0 + 31 > rowbase) {   // causal mask near the diagonal
#pragma unroll
                for (int r = 0; r < 8; ++r) {
                    int row = rowbase + r + 8 * g;
                    int c0 = k0 + lane16;
                    if (c0 > row)      s[m][0][r] = -10000.0f;
                    if (c0 + 16 > row) s[m][1][r] = -10000.0f;
                }
            }
            // online softmax per row (rows live in 16-lane groups)
#pragma unroll
            for (int r = 0; r < 8; ++r) {
                float mb = fmaxf(s[m][0][r], s[m][1][r]);
                mb = fmaxf(mb, __shfl_xor(mb, 1));
                mb = fmaxf(mb, __shfl_xor(mb, 2));
                mb = fmaxf(mb, __shfl_xor(mb, 4));
                mb = fmaxf(mb, __shfl_xor(mb, 8));
                float mnew  = fmaxf(m_run[m][r], mb);
                float alpha = __expf(m_run[m][r] - mnew);
                m_run[m][r] = mnew;
                float p0 = __expf(s[m][0][r] - mnew);
                float p1 = __expf(s[m][1][r] - mnew);
                float rs = p0 + p1;
                rs += __shfl_xor(rs, 1);
                rs += __shfl_xor(rs, 2);
                rs += __shfl_xor(rs, 4);
                rs += __shfl_xor(rs, 8);
                l_run[m][r] = l_run[m][r] * alpha + rs;
                acc[m][0][r] *= alpha; acc[m][1][r] *= alpha;
                acc[m][2][r] *= alpha; acc[m][3][r] *= alpha;
                int mr = m * 16 + r + 8 * g;
                pw[mr * 40 + lane16]      = f2bfu(p0);
                pw[mr * 40 + 16 + lane16] = f2bfu(p1);
            }
        }

        // ---- phase 3: reload P in A-layout, accumulate P @ V ----
        v16bf PA[2];
#pragma unroll
        for (int m = 0; m < 2; ++m) {
            int mr = m * 16 + lane16;
        PA[m] = cat8(*(const v8bf*)&pw[mr * 40 + g * 16],
                     *(const v8bf*)&pw[mr * 40 + g * 16 + 8]);
        }
#pragma unroll
        for (int m = 0; m < 2; ++m)
#pragma unroll
            for (int j = 0; j < 4; ++j)
                acc[m][j] = wmma_bf16(PA[m], VB[j], acc[m][j]);
    }

    // normalize and store ctx as bf16 [B, S, H*Dh]
    int b = bh >> 4, h = bh & 15;
#pragma unroll
    for (int m = 0; m < 2; ++m)
#pragma unroll
        for (int r = 0; r < 8; ++r) {
            float inv = 1.0f / l_run[m][r];
            int row = b * S_LEN + qbase + m * 16 + r + 8 * g;
#pragma unroll
            for (int j = 0; j < 4; ++j) {
                int col = h * 64 + j * 16 + lane16;
                ctxws[(size_t)row * D_MODEL + col] = f2bfu(acc[m][j][r] * inv);
            }
        }
}

// ---------------------------------------------------------------------------
// Output projection: out = ctx @ Wo^T + bo (bf16 x bf16 -> f32), 32x64/wave.
// ---------------------------------------------------------------------------
__global__ void __launch_bounds__(256)
outproj_kernel(const unsigned short* __restrict__ ctx,
               const unsigned short* __restrict__ Wobf,
               const float* __restrict__ bo,
               float* __restrict__ out) {
    int tid = blockIdx.x * blockDim.x + threadIdx.x;
    int w = tid >> 5;
    int lane = tid & 31, lane16 = lane & 15, g = lane >> 4;
    int mtile = w >> 4;          // 0..255
    int jt = w & 15;             // 64-col block

    const unsigned short* abase = ctx  + (size_t)(mtile * 32 + lane16) * D_MODEL;
    const unsigned short* bbase = Wobf + (size_t)(jt * 64 + lane16) * D_MODEL + g * 16;

    v8f z = {0.f,0.f,0.f,0.f,0.f,0.f,0.f,0.f};
    v8f acc[2][4];
#pragma unroll
    for (int m = 0; m < 2; ++m)
#pragma unroll
        for (int j = 0; j < 4; ++j) acc[m][j] = z;

    v16bf A0[2], B0[4], A1[2], B1[4];
    load_a2(A0, abase, 0, g);
    load_b4(B0, bbase, 0);
    for (int d0 = 0; d0 < D_MODEL; d0 += 64) {
        load_a2(A1, abase, d0 + 32, g);
        load_b4(B1, bbase, d0 + 32);
        mma8(acc, A0, B0);
        if (d0 + 64 < D_MODEL) {
            load_a2(A0, abase, d0 + 64, g);
            load_b4(B0, bbase, d0 + 64);
        }
        mma8(acc, A1, B1);
    }

#pragma unroll
    for (int j = 0; j < 4; ++j) {
        int col = jt * 64 + j * 16 + lane16;
        float bv = bo[col];
#pragma unroll
        for (int m = 0; m < 2; ++m)
#pragma unroll
            for (int r = 0; r < 8; ++r) {
                int row = mtile * 32 + m * 16 + r + 8 * g;
                out[(size_t)row * D_MODEL + col] = acc[m][j][r] + bv;
            }
    }
}

// ---------------------------------------------------------------------------
// Launcher. Workspace layout (ushorts):
//   [0,8M)     xbf (reused as ctx)      [8M,12M) Wq/Wk/Wv/Wo bf16
//   [12M,20M)  Q bf16 [b,h,s,d]         [20M,28M) K bf16 [b,h,s,d]
//   [28M,36M)  V bf16 transposed [b,h,d,s]        total 72 MB
// ---------------------------------------------------------------------------
extern "C" void kernel_launch(void* const* d_in, const int* in_sizes, int n_in,
                              void* d_out, int out_size, void* d_ws, size_t ws_size,
                              hipStream_t stream) {
    (void)in_sizes; (void)n_in; (void)out_size; (void)ws_size;
    const float* x  = (const float*)d_in[0];
    // d_in[1] = mask (causal, handled analytically)
    const float* Wq = (const float*)d_in[2];
    const float* bq = (const float*)d_in[3];
    const float* Wk = (const float*)d_in[4];
    const float* bk = (const float*)d_in[5];
    const float* Wv = (const float*)d_in[6];
    const float* bv = (const float*)d_in[7];
    const float* Wo = (const float*)d_in[8];
    const float* bo = (const float*)d_in[9];
    float* out = (float*)d_out;

    const size_t XN = (size_t)NB * S_LEN * D_MODEL;       // 8388608
    const size_t WN = (size_t)D_MODEL * D_MODEL;          // 1048576
    unsigned short* ws   = (unsigned short*)d_ws;
    unsigned short* xbf  = ws;
    unsigned short* wqb  = ws + XN;
    unsigned short* wkb  = wqb + WN;
    unsigned short* wvb  = wkb + WN;
    unsigned short* wob  = wvb + WN;
    unsigned short* qws  = wob + WN;
    unsigned short* kws  = qws + XN;
    unsigned short* vtws = kws + XN;
    unsigned short* ctxw = xbf;                            // reuse x buffer

    dim3 blk(256);
    cvt_bf16_kernel<<<(int)(XN / 4 / 256), blk, 0, stream>>>(x,  xbf, (int)(XN / 4));
    cvt_bf16_kernel<<<(int)(WN / 4 / 256), blk, 0, stream>>>(Wq, wqb, (int)(WN / 4));
    cvt_bf16_kernel<<<(int)(WN / 4 / 256), blk, 0, stream>>>(Wk, wkb, (int)(WN / 4));
    cvt_bf16_kernel<<<(int)(WN / 4 / 256), blk, 0, stream>>>(Wv, wvb, (int)(WN / 4));
    cvt_bf16_kernel<<<(int)(WN / 4 / 256), blk, 0, stream>>>(Wo, wob, (int)(WN / 4));

    // 256 mtiles(32 rows) * 16 heads = 4096 waves = 512 blocks of 8 waves
    proj_kernel<<<512, blk, 0, stream>>>(xbf, wqb, bq, qws, 0);
    proj_kernel<<<512, blk, 0, stream>>>(xbf, wkb, bk, kws, 1);
    proj_kernel<<<512, blk, 0, stream>>>(xbf, wvb, bv, vtws, 2);

    // 64 (b,h) * 64 q-tiles(32 rows) = 4096 waves = 512 blocks
    attn_kernel<<<512, blk, 0, stream>>>(qws, kws, vtws, ctxw);

    // 256 mtiles * 16 col-blocks = 4096 waves = 512 blocks
    outproj_kernel<<<512, blk, 0, stream>>>(ctxw, wob, bo, out);
}
